// TransformerEncoderLayer_44659069944108
// MI455X (gfx1250) — matrix operand order repaired and checked by reading, not compiled, not run on hardware
//
#include <hip/hip_runtime.h>
#include <hip/hip_bf16.h>

typedef _Float16 half_t;
typedef __attribute__((ext_vector_type(16))) _Float16 v16h;
typedef __attribute__((ext_vector_type(8)))  _Float16 v8h;
typedef __attribute__((ext_vector_type(8)))  float    v8f;

#define D_MODEL 1024
#define HEADS   16
#define HEAD_DIM 64
#define HIDDEN  4096
#define SEQ     2048
#define BATCH   2
#define MTOK    4096   // BATCH*SEQ
#define LN_EPS  1e-5f

// ---------------------------------------------------------------------------
// WMMA helper:  D = A(16x32 f16) * B(32x16 f16) + C(16x16 f32)
// ---------------------------------------------------------------------------
static __device__ __forceinline__ v8f wmma_f16(v16h a, v16h b, v8f c) {
  return __builtin_amdgcn_wmma_f32_16x16x32_f16(false, a, false, b, (short)0, c,
                                                false, false);
}

// A-fragment (16 rows x 32 k), row-major source, k contiguous.
// ISA layout: lane l -> row l%16; k chunks {c0..c0+7, c0+16..c0+23}, c0=8*(l/16).
static __device__ __forceinline__ v16h load_frag_a(const half_t* __restrict__ base, int ld) {
  int l = threadIdx.x & 31;
  const half_t* p = base + (size_t)(l & 15) * ld + ((l >> 4) << 3);
  v8h lo = *(const v8h*)p;
  v8h hi = *(const v8h*)(p + 16);
  v16h r;
#pragma unroll
  for (int i = 0; i < 8; ++i) { r[i] = lo[i]; r[8 + i] = hi[i]; }
  return r;
}

// B-fragment (32 k x 16 cols). Source is the TRANSPOSE of B (row n = column n
// of B, k contiguous). ISA layout: lane l -> col l%16; k chunks {c0..c0+7, c0+16..c0+23}, c0=8*(l/16).
static __device__ __forceinline__ v16h load_frag_b(const half_t* __restrict__ base, int ld) {
  int l = threadIdx.x & 31;
  const half_t* p = base + (size_t)(l & 15) * ld + ((l >> 4) << 3);
  v8h lo = *(const v8h*)p;
  v8h hi = *(const v8h*)(p + 16);
  v16h r;
#pragma unroll
  for (int i = 0; i < 8; ++i) { r[i] = lo[i]; r[8 + i] = hi[i]; }
  return r;
}

// ---------------------------------------------------------------------------
// Elementwise fp32 -> f16
// ---------------------------------------------------------------------------
__global__ void cvt_f32_to_f16(const float* __restrict__ in, half_t* __restrict__ out, int n) {
  int i = blockIdx.x * blockDim.x + threadIdx.x;
  if (i < n) out[i] = (half_t)in[i];
}

// ---------------------------------------------------------------------------
// Transpose+convert: in f32 [R][C]  ->  out f16 [C][R]
// ---------------------------------------------------------------------------
__global__ void __launch_bounds__(256)
transpose_cvt(const float* __restrict__ in, half_t* __restrict__ out, int R, int C) {
  __shared__ float tile[32][33];
  int tx = threadIdx.x, ty = threadIdx.y;
  int r0 = blockIdx.y * 32, c0 = blockIdx.x * 32;
#pragma unroll
  for (int j = 0; j < 4; ++j)
    tile[ty + j * 8][tx] = in[(size_t)(r0 + ty + j * 8) * C + (c0 + tx)];
  __syncthreads();
#pragma unroll
  for (int j = 0; j < 4; ++j)
    out[(size_t)(c0 + ty + j * 8) * R + (r0 + tx)] = (half_t)tile[tx][ty + j * 8];
}

// ---------------------------------------------------------------------------
// WMMA GEMM:  C[M][N] = A[M][K](f16) @ W[K][N] + bias, W given transposed
// WT[N][K](f16).  256 threads = 8 waves arranged 2(M) x 4(N); block tile
// 64(M) x 256(N); per-wave tile 32x64 (8 accumulators, A-frags reused over
// 4 B-frags -> 8 WMMA per 12 b128 loads per lane).
// MODE 0: f32 out.  MODE 1: f16 ReLU out.  MODE 2: qkv scatter (q,k head-major;
//         v transposed [b,h,d,L]).
// ---------------------------------------------------------------------------
template <int MODE>
__global__ void __launch_bounds__(256)
gemm_wmma(const half_t* __restrict__ A, const half_t* __restrict__ WT,
          const float* __restrict__ bias,
          float* __restrict__ outF, half_t* __restrict__ outH,
          half_t* __restrict__ qh, half_t* __restrict__ kh, half_t* __restrict__ vt,
          int M, int N, int K) {
  int wave = threadIdx.x >> 5;
  int lane = threadIdx.x & 31;
  int m0 = blockIdx.y * 64 + (wave & 1) * 32;
  int n0 = blockIdx.x * 256 + (wave >> 1) * 64;

  v8f acc[2][4] = {};
  const half_t* Ar0 = A + (size_t)m0 * K;
  const half_t* Ar1 = A + (size_t)(m0 + 16) * K;
  const half_t* Br[4] = {
    WT + (size_t)(n0 +  0) * K,
    WT + (size_t)(n0 + 16) * K,
    WT + (size_t)(n0 + 32) * K,
    WT + (size_t)(n0 + 48) * K
  };

  for (int k0 = 0; k0 < K; k0 += 32) {
    v16h a0 = load_frag_a(Ar0 + k0, K);
    v16h a1 = load_frag_a(Ar1 + k0, K);
#pragma unroll
    for (int tj = 0; tj < 4; ++tj) {
      v16h b = load_frag_b(Br[tj] + k0, K);
      acc[0][tj] = wmma_f16(a0, b, acc[0][tj]);
      acc[1][tj] = wmma_f16(a1, b, acc[1][tj]);
    }
  }

  int col = lane & 15;
  int rg  = lane >> 4;
#pragma unroll
  for (int ti = 0; ti < 2; ++ti) {
#pragma unroll
    for (int tj = 0; tj < 4; ++tj) {
      int n = n0 + tj * 16 + col;
      float bv = bias[n];
#pragma unroll
      for (int r = 0; r < 8; ++r) {
        int m = m0 + ti * 16 + r + 8 * rg;
        float val = acc[ti][tj][r] + bv;
        if (MODE == 0) {
          outF[(size_t)m * N + n] = val;
        } else if (MODE == 1) {
          outH[(size_t)m * N + n] = (half_t)fmaxf(val, 0.0f);
        } else {
          int s = n >> 10, h = (n >> 6) & 15, d = n & 63;
          int b = m >> 11, t = m & (SEQ - 1);
          size_t bh = (size_t)(b * HEADS + h);
          if (s == 0)      qh[(bh * SEQ + t) * HEAD_DIM + d] = (half_t)val;
          else if (s == 1) kh[(bh * SEQ + t) * HEAD_DIM + d] = (half_t)val;
          else             vt[(bh * HEAD_DIM + d) * SEQ + t] = (half_t)val;
        }
      }
    }
  }
}

// ---------------------------------------------------------------------------
// Flash attention. One wave handles 16 query rows of one (b,h); keys consumed
// 32 at a time.  Trick: compute T = K·Qᵀ so the score tile's C-layout (lane =
// query column, regs = key rows) IS the A-fragment layout for the P·V WMMA —
// no cross-lane data movement between softmax and the PV matmul.
// ---------------------------------------------------------------------------
__global__ void __launch_bounds__(256)
attn_flash(const half_t* __restrict__ q, const half_t* __restrict__ k,
           const half_t* __restrict__ vt, half_t* __restrict__ ctx) {
  int lane = threadIdx.x & 31;
  int gw = blockIdx.x * (blockDim.x >> 5) + (threadIdx.x >> 5);
  int bh = gw >> 7;            // 128 row-blocks of 16 per (b,h)
  int rb = gw & 127;
  int b  = bh >> 4;
  int h  = bh & 15;

  const half_t* Qb = q  + ((size_t)bh * SEQ + rb * 16) * HEAD_DIM;
  const half_t* Kb = k  + (size_t)bh * SEQ * HEAD_DIM;
  const half_t* Vb = vt + (size_t)bh * HEAD_DIM * SEQ;

  // Q as WMMA-B operand for T = K·Qᵀ (contraction over head_dim).
  v16h qb0 = load_frag_b(Qb, HEAD_DIM);        // head dims 0..31
  v16h qb1 = load_frag_b(Qb + 32, HEAD_DIM);   // head dims 32..63

  float mrun = -3.0e38f, lrun = 0.0f;
  v8f o0 = {}, o1 = {}, o2 = {}, o3 = {};
  const float scale = 0.125f;                  // 1/sqrt(64)

  for (int kb = 0; kb < SEQ; kb += 32) {
    v16h a00 = load_frag_a(Kb + (size_t)kb * HEAD_DIM, HEAD_DIM);
    v16h a01 = load_frag_a(Kb + (size_t)kb * HEAD_DIM + 32, HEAD_DIM);
    v16h a10 = load_frag_a(Kb + (size_t)(kb + 16) * HEAD_DIM, HEAD_DIM);
    v16h a11 = load_frag_a(Kb + (size_t)(kb + 16) * HEAD_DIM + 32, HEAD_DIM);
    v8f z = {};
    v8f t0 = wmma_f16(a00, qb0, z);  t0 = wmma_f16(a01, qb1, t0); // keys kb..+15
    v8f t1 = wmma_f16(a10, qb0, z);  t1 = wmma_f16(a11, qb1, t1); // keys kb+16..+31

    float bm = -3.0e38f;
#pragma unroll
    for (int i = 0; i < 8; ++i) {
      t0[i] *= scale; t1[i] *= scale;
      bm = fmaxf(bm, fmaxf(t0[i], t1[i]));
    }
    bm = fmaxf(bm, __shfl_xor(bm, 16, 32));    // other 16 key rows live in lane^16
    float mnew = fmaxf(mrun, bm);
    float corr = __expf(mrun - mnew);
    float bs = 0.0f;
#pragma unroll
    for (int i = 0; i < 8; ++i) {
      t0[i] = __expf(t0[i] - mnew);
      t1[i] = __expf(t1[i] - mnew);
      bs += t0[i] + t1[i];
    }
    bs += __shfl_xor(bs, 16, 32);
    lrun = lrun * corr + bs;
    mrun = mnew;

    // Rescale running O: row q of O is reg r at lane-group l/16 (q = r+8*(l/16));
    // corr for query q sits in lanes q and q+16.
#pragma unroll
    for (int r = 0; r < 8; ++r) {
      float c = __shfl(corr, r + 8 * (lane >> 4), 32);
      o0[r] *= c; o1[r] *= c; o2[r] *= c; o3[r] *= c;
    }

    // P as A-fragment: direct per-lane repack (C-layout of T == A-layout of P).
    v16h pa;
#pragma unroll
    for (int i = 0; i < 8; ++i) { pa[i] = (half_t)t0[i]; pa[8 + i] = (half_t)t1[i]; }

    v16h vb0 = load_frag_b(Vb + (size_t)0  * SEQ + kb, SEQ);
    v16h vb1 = load_frag_b(Vb + (size_t)16 * SEQ + kb, SEQ);
    v16h vb2 = load_frag_b(Vb + (size_t)32 * SEQ + kb, SEQ);
    v16h vb3 = load_frag_b(Vb + (size_t)48 * SEQ + kb, SEQ);
    o0 = wmma_f16(pa, vb0, o0);
    o1 = wmma_f16(pa, vb1, o1);
    o2 = wmma_f16(pa, vb2, o2);
    o3 = wmma_f16(pa, vb3, o3);
  }

  float linv = 1.0f / lrun;
#pragma unroll
  for (int r = 0; r < 8; ++r) {
    float c = __shfl(linv, r + 8 * (lane >> 4), 32);
    int t = rb * 16 + r + 8 * (lane >> 4);
    size_t base = ((size_t)(b * SEQ + t)) * D_MODEL + h * HEAD_DIM + (lane & 15);
    ctx[base + 0]  = (half_t)(o0[r] * c);
    ctx[base + 16] = (half_t)(o1[r] * c);
    ctx[base + 32] = (half_t)(o2[r] * c);
    ctx[base + 48] = (half_t)(o3[r] * c);
  }
}

// ---------------------------------------------------------------------------
// Residual add + LayerNorm over D_MODEL=1024; one block per row.
// Writes f32 (and optional f16 copy for the next GEMM).
// ---------------------------------------------------------------------------
__global__ void __launch_bounds__(256)
ln_residual(const float* __restrict__ a, const float* __restrict__ r2,
            const float* __restrict__ g, const float* __restrict__ be,
            float* __restrict__ outF, half_t* __restrict__ outH) {
  int row = blockIdx.x;
  int tid = threadIdx.x;
  size_t base = (size_t)row * D_MODEL;
  float vals[4];
  float s = 0.0f, ss = 0.0f;
#pragma unroll
  for (int j = 0; j < 4; ++j) {
    int i = tid + j * 256;
    float v = a[base + i] + r2[base + i];
    vals[j] = v; s += v; ss += v * v;
  }
#pragma unroll
  for (int off = 16; off > 0; off >>= 1) {
    s  += __shfl_xor(s, off, 32);
    ss += __shfl_xor(ss, off, 32);
  }
  __shared__ float sm[8], sm2[8];
  if ((tid & 31) == 0) { sm[tid >> 5] = s; sm2[tid >> 5] = ss; }
  __syncthreads();
  float ts = 0.0f, tss = 0.0f;
#pragma unroll
  for (int w = 0; w < 8; ++w) { ts += sm[w]; tss += sm2[w]; }
  float mu  = ts  * (1.0f / D_MODEL);
  float var = tss * (1.0f / D_MODEL) - mu * mu;
  float rs  = rsqrtf(var + LN_EPS);
#pragma unroll
  for (int j = 0; j < 4; ++j) {
    int i = tid + j * 256;
    float y = (vals[j] - mu) * rs * g[i] + be[i];
    outF[base + i] = y;
    if (outH) outH[base + i] = (half_t)y;
  }
}

// ---------------------------------------------------------------------------
// Host orchestration
// ---------------------------------------------------------------------------
extern "C" void kernel_launch(void* const* d_in, const int* in_sizes, int n_in,
                              void* d_out, int out_size, void* d_ws, size_t ws_size,
                              hipStream_t stream) {
  (void)in_sizes; (void)n_in; (void)out_size; (void)ws_size;
  const float* x     = (const float*)d_in[0];
  const float* w_qkv = (const float*)d_in[1];
  const float* b_qkv = (const float*)d_in[2];
  const float* w_out = (const float*)d_in[3];
  const float* b_out = (const float*)d_in[4];
  const float* w1    = (const float*)d_in[5];
  const float* b1    = (const float*)d_in[6];
  const float* w2    = (const float*)d_in[7];
  const float* b2    = (const float*)d_in[8];
  const float* g1    = (const float*)d_in[9];
  const float* be1   = (const float*)d_in[10];
  const float* g2    = (const float*)d_in[11];
  const float* be2   = (const float*)d_in[12];

  char* ws = (char*)d_ws;
  const size_t MB = 1048576;
  // Phase-aliased workspace layout (120 MiB total):
  half_t* x_h     = (half_t*)(ws + 0 * MB);    // dead after qkv GEMM
  half_t* w1T     = (half_t*)(ws + 0 * MB);    // reuses x_h region
  half_t* wqkvT   = (half_t*)(ws + 8 * MB);    // dead after qkv GEMM
  half_t* w2T     = (half_t*)(ws + 8 * MB);    // reuses wqkvT+woutT region
  half_t* woutT   = (half_t*)(ws + 14 * MB);   // dead after out-proj GEMM
  half_t* q_h     = (half_t*)(ws + 16 * MB);
  half_t* k_h     = (half_t*)(ws + 24 * MB);
  half_t* v_t     = (half_t*)(ws + 32 * MB);
  half_t* ctx_h   = (half_t*)(ws + 40 * MB);
  float*  attn_o  = (float*) (ws + 48 * MB);   // dead after LN1
  float*  ffn2    = (float*) (ws + 48 * MB);   // reuses attn_o region
  float*  h_f32   = (float*) (ws + 64 * MB);
  half_t* h_h     = (half_t*)(ws + 80 * MB);
  half_t* ffn1_h  = (half_t*)(ws + 88 * MB);

  dim3 tb(32, 8);

  // 1) x -> f16
  cvt_f32_to_f16<<<(MTOK * D_MODEL) / 256, 256, 0, stream>>>(x, x_h, MTOK * D_MODEL);
  // 2) w_qkv [1024][3072] -> wqkvT [3072][1024]
  transpose_cvt<<<dim3(3072 / 32, 1024 / 32), tb, 0, stream>>>(w_qkv, wqkvT, 1024, 3072);
  // 3) qkv GEMM + scatter to q,k (head-major) and v (transposed)
  gemm_wmma<2><<<dim3(3072 / 256, MTOK / 64), 256, 0, stream>>>(
      x_h, wqkvT, b_qkv, nullptr, nullptr, q_h, k_h, v_t, MTOK, 3072, D_MODEL);
  // 4) flash attention -> ctx (token-major f16)
  attn_flash<<<512, 256, 0, stream>>>(q_h, k_h, v_t, ctx_h);
  // 5) w_out -> woutT; out-projection GEMM -> attn_o (f32)
  transpose_cvt<<<dim3(1024 / 32, 1024 / 32), tb, 0, stream>>>(w_out, woutT, 1024, 1024);
  gemm_wmma<0><<<dim3(1024 / 256, MTOK / 64), 256, 0, stream>>>(
      ctx_h, woutT, b_out, attn_o, nullptr, nullptr, nullptr, nullptr, MTOK, 1024, D_MODEL);
  // 6) h = LN(x + attn_o)
  ln_residual<<<MTOK, 256, 0, stream>>>(x, attn_o, g1, be1, h_f32, h_h);
  // 7) w1 [1024][4096] -> w1T [4096][1024]; FFN GEMM1 + ReLU -> f16
  transpose_cvt<<<dim3(4096 / 32, 1024 / 32), tb, 0, stream>>>(w1, w1T, 1024, 4096);
  gemm_wmma<1><<<dim3(4096 / 256, MTOK / 64), 256, 0, stream>>>(
      h_h, w1T, b1, nullptr, ffn1_h, nullptr, nullptr, nullptr, MTOK, HIDDEN, D_MODEL);
  // 8) w2 [4096][1024] -> w2T [1024][4096]; FFN GEMM2 -> f32
  transpose_cvt<<<dim3(1024 / 32, 4096 / 32), tb, 0, stream>>>(w2, w2T, 4096, 1024);
  gemm_wmma<0><<<dim3(1024 / 256, MTOK / 64), 256, 0, stream>>>(
      ffn1_h, w2T, b2, ffn2, nullptr, nullptr, nullptr, nullptr, MTOK, D_MODEL, HIDDEN);
  // 9) out = LN(h + ffn2)
  ln_residual<<<MTOK, 256, 0, stream>>>(h_f32, ffn2, g2, be2, (float*)d_out, nullptr);
}
